// GraphTransformerPooling_60979945669317
// MI455X (gfx1250) — matrix operand order repaired and checked by reading, not compiled, not run on hardware
//
#include <hip/hip_runtime.h>
#include <hip/hip_bf16.h>
#include <math.h>

// ---------------------------------------------------------------------------
// GraphTransformerPooling, MI455X (gfx1250, wave32, WMMA).
//
// K-GEMM folded into H=4 score vectors; V-GEMM folded to the right of the
// segment reduction. One streaming pass over node_feat (128 MB -> ~5.5us
// at 23.3 TB/s) with online softmax and WMMA bf16 rank-32 updates; then
// three tiny WMMA GEMMs. This revision replaces __syncthreads() in the hot
// loop with an LDS-only split barrier (s_wait_dscnt 0 + s_barrier_signal/
// wait): HIP's __syncthreads fence was draining loadcnt to 0 at every
// barrier, killing the ping-pong global-load pipeline. Only DS visibility
// is required here; global loads are private register dataflow.
// ---------------------------------------------------------------------------

typedef __attribute__((ext_vector_type(16))) __bf16 v16bf;
typedef __attribute__((ext_vector_type(8)))  float  v8f;

#define NN   131072
#define DD   256
#define BB   256
#define HH   4
#define DH   64
#define OUTD 512

// Workgroup barrier with LDS-only visibility: each wave drains its own
// outstanding DS ops (in-order per wave) before signaling, so all LDS
// writes are visible after the wait. Deliberately does NOT drain loadcnt.
__device__ __forceinline__ void wg_barrier_lds() {
    asm volatile("s_wait_dscnt 0x0\n\t"
                 "s_barrier_signal -1\n\t"
                 "s_barrier_wait -1" ::: "memory");
}

__device__ __forceinline__ int lower_bound_i32(const int* __restrict__ a, int n, int key) {
    int lo = 0, hi = n;
    while (lo < hi) {
        int mid = (lo + hi) >> 1;
        if (a[mid] < key) lo = mid + 1; else hi = mid;
    }
    return lo;
}

// ---------------------------------------------------------------------------
// Kernel 1 (tiny, 1 block): q = (gq @ Wq^T + bq)*scale;
// qk[h][j] = sum_d q[h,d]*Wk[h*64+d, j]; sb[h] = q_h . bk_h
// ---------------------------------------------------------------------------
__global__ void __launch_bounds__(256)
setup_kernel(const float* __restrict__ gq, const float* __restrict__ in_w,
             const float* __restrict__ in_b, float* __restrict__ qk,
             float* __restrict__ sb)
{
    __shared__ float s_q[DD];
    const int t = threadIdx.x;
    float acc = in_b[t];
    const float* wrow = in_w + (size_t)t * DD;            // Wq row t
    for (int j = 0; j < DD; ++j) acc += gq[j] * wrow[j];
    s_q[t] = acc * 0.125f;                                // 1/sqrt(DH=64)
    __syncthreads();
    for (int h = 0; h < HH; ++h) {
        float a = 0.f;
        for (int d = 0; d < DH; ++d)
            a += s_q[h * DH + d] * in_w[(size_t)(DD + h * DH + d) * DD + t]; // Wk
        qk[h * DD + t] = a;
    }
    if (t < HH) {
        float a = 0.f;
        for (int d = 0; d < DH; ++d) a += s_q[t * DH + d] * in_b[DD + t * DH + d]; // bk
        sb[t] = a;
    }
}

// ---------------------------------------------------------------------------
// Kernel 2 helpers
// ---------------------------------------------------------------------------
__device__ __forceinline__ void load_chunk(float4 (&x)[8],
                                           const float* __restrict__ node_feat,
                                           int base, int end, int row, int dbase)
{
    const int  node = base + row;
    const int  cn   = (node < end) ? node : (end - 1);    // clamp: dup rows get e=0
    const float* src = node_feat + (size_t)cn * DD + dbase;
    #pragma unroll
    for (int i = 0; i < 8; ++i) x[i] = ((const float4*)src)[i];
    const int pn = (node + 64 < NN) ? (node + 64) : (NN - 1); // GL2 warm, in-bounds
    __builtin_prefetch(node_feat + (size_t)pn * DD + dbase, 0, 0);
}

__device__ __forceinline__ void process_chunk(const float4 (&x)[8], int base, int end,
    int row, int seg, int dbase, int lane, int wv, int jbase,
    __bf16 (*s_tile)[DD + 8], float (*s_qk)[DD], float (*s_ps)[HH][33],
    __bf16 (*s_e)[HH], float* s_factor, const float* s_sb,
    v8f& c0, v8f& c1, float& m_run, float& den_run)
{
    wg_barrier_lds();
    // ---- phase 1: fp32 -> bf16 LDS tile + fused score partials
    {
        float p0 = 0.f, p1 = 0.f, p2 = 0.f, p3 = 0.f;
        #pragma unroll
        for (int i = 0; i < 8; ++i) {
            const int d = dbase + i * 4;
            s_tile[row][d + 0] = (__bf16)x[i].x;
            s_tile[row][d + 1] = (__bf16)x[i].y;
            s_tile[row][d + 2] = (__bf16)x[i].z;
            s_tile[row][d + 3] = (__bf16)x[i].w;
            p0 += x[i].x * s_qk[0][d] + x[i].y * s_qk[0][d+1]
                + x[i].z * s_qk[0][d+2] + x[i].w * s_qk[0][d+3];
            p1 += x[i].x * s_qk[1][d] + x[i].y * s_qk[1][d+1]
                + x[i].z * s_qk[1][d+2] + x[i].w * s_qk[1][d+3];
            p2 += x[i].x * s_qk[2][d] + x[i].y * s_qk[2][d+1]
                + x[i].z * s_qk[2][d+2] + x[i].w * s_qk[2][d+3];
            p3 += x[i].x * s_qk[3][d] + x[i].y * s_qk[3][d+1]
                + x[i].z * s_qk[3][d+2] + x[i].w * s_qk[3][d+3];
        }
        s_ps[seg][0][row] = p0; s_ps[seg][1][row] = p1;
        s_ps[seg][2][row] = p2; s_ps[seg][3][row] = p3;
    }
    wg_barrier_lds();
    // ---- phase 2: waves 0..3 (head = wv, node = lane): online softmax
    if (wv < HH) {
        const int h = wv, n = lane;
        float score = s_sb[h];
        #pragma unroll
        for (int g = 0; g < 8; ++g) score += s_ps[g][h][n];
        if (base + n >= end) score = -1e30f;
        float mx = score;
        #pragma unroll
        for (int off = 16; off > 0; off >>= 1)
            mx = fmaxf(mx, __shfl_xor(mx, off, 32));
        const float nm  = fmaxf(m_run, mx);
        const float fac = __expf(m_run - nm);
        m_run = nm;
        const float e = __expf(score - nm);
        s_e[n][h] = (__bf16)e;
        float es = e;
        #pragma unroll
        for (int off = 16; off > 0; off >>= 1)
            es += __shfl_xor(es, off, 32);
        den_run = den_run * fac + es;
        if (lane == 0) s_factor[h] = fac;
    }
    wg_barrier_lds();
    // ---- phase 3: rescale accumulators + WMMA rank-32 accumulation.
    // C rows 0..3 (heads) sit in VGPRs 0..3 of lanes 0-15; rows >=4 are zero.
    {
        const float f0 = s_factor[0], f1 = s_factor[1];
        const float f2 = s_factor[2], f3 = s_factor[3];
        c0[0] *= f0; c0[1] *= f1; c0[2] *= f2; c0[3] *= f3;
        c1[0] *= f0; c1[1] *= f1; c1[2] *= f2; c1[3] *= f3;
    }
    v16bf a;
    {
        const int m = lane & 15, half = lane >> 4;
        #pragma unroll
        for (int i = 0; i < 8; ++i) {
            const int k0 = half * 16 + i;
            const int k1 = half * 16 + 8 + i;
            a[i]     = (m < HH) ? s_e[k0][m] : (__bf16)0.0f;
            a[8 + i] = (m < HH) ? s_e[k1][m] : (__bf16)0.0f;
        }
    }
    {
        const int n  = lane & 15;
        const int ko = (lane >> 4) * 16;
        v16bf bf;
        #pragma unroll
        for (int i = 0; i < 16; ++i) bf[i] = s_tile[ko + i][jbase + n];
        c0 = __builtin_amdgcn_wmma_f32_16x16x32_bf16(false, a, false, bf,
                                                     (short)0, c0, false, false);
        #pragma unroll
        for (int i = 0; i < 16; ++i) bf[i] = s_tile[ko + i][jbase + 16 + n];
        c1 = __builtin_amdgcn_wmma_f32_16x16x32_bf16(false, a, false, bf,
                                                     (short)0, c1, false, false);
    }
}

// ---------------------------------------------------------------------------
// Kernel 2: one workgroup (8 wave32) per graph segment.
// ---------------------------------------------------------------------------
__global__ void __launch_bounds__(256)
segment_kernel(const float* __restrict__ node_feat, const int* __restrict__ seg_ids,
               const float* __restrict__ qk_ws, const float* __restrict__ sb_ws,
               float* __restrict__ Enorm)
{
    __shared__ __bf16 s_tile[32][DD + 8];     // 32-node tile, bf16, padded
    __shared__ float  s_qk[HH][DD];
    __shared__ float  s_ps[8][HH][33];        // [seg][h][node], conflict-free
    __shared__ __bf16 s_e[32][HH];
    __shared__ float  s_factor[HH], s_invden[HH], s_sb[HH];

    const int t     = threadIdx.x;
    const int b     = blockIdx.x;
    const int lane  = t & 31;
    const int wv    = t >> 5;
    const int jbase = wv * 32;                // this wave owns columns [jbase, jbase+32)

    for (int j = t; j < HH * DD; j += 256) ((float*)s_qk)[j] = qk_ws[j];
    if (t < HH) s_sb[t] = sb_ws[t];

    const int start = lower_bound_i32(seg_ids, NN, b);
    const int end   = lower_bound_i32(seg_ids, NN, b + 1);

    v8f c0 = {}; v8f c1 = {};                 // E accumulators (rows 0..3 = heads)
    float m_run = -1e30f, den_run = 0.f;      // live in waves 0..3 (head = wv)

    const int row   = t >> 3;                 // node within chunk (0..31)
    const int seg   = t & 7;                  // 32-dim slice within row
    const int dbase = seg * 32;

    // ---- ping-pong pipeline over two named buffers
    float4 xa[8], xb[8];
    if (start < end) load_chunk(xa, node_feat, start, end, row, dbase);

    for (int base = start; base < end; base += 64) {
        load_chunk(xb, node_feat, base + 32, end, row, dbase);   // stage B loads
        process_chunk(xa, base, end, row, seg, dbase, lane, wv, jbase,
                      s_tile, s_qk, s_ps, s_e, s_factor, s_sb,
                      c0, c1, m_run, den_run);
        load_chunk(xa, node_feat, base + 64, end, row, dbase);   // next A loads
        if (base + 32 < end)                                      // block-uniform
            process_chunk(xb, base + 32, end, row, seg, dbase, lane, wv, jbase,
                          s_tile, s_qk, s_ps, s_e, s_factor, s_sb,
                          c0, c1, m_run, den_run);
    }

    if (wv < HH && lane == 0)
        s_invden[wv] = (den_run > 0.f) ? (1.0f / den_run) : 0.f;
    wg_barrier_lds();
    // C layout: lanes 0-15 hold rows 0..7 (we need rows 0..3 = heads), N = lane.
    if (lane < 16) {
        float* dst = Enorm + (size_t)b * (HH * DD);
        #pragma unroll
        for (int h = 0; h < HH; ++h) {
            dst[h * DD + jbase + lane]      = c0[h] * s_invden[h];
            dst[h * DD + jbase + 16 + lane] = c1[h] * s_invden[h];
        }
    }
}

// ---------------------------------------------------------------------------
// Kernel 3: pooled[b, h*64+d] = Enorm[b,h,:] . Wv[h*64+d,:] + bv[h*64+d]
// ---------------------------------------------------------------------------
__global__ void __launch_bounds__(128)
pooled_kernel(const float* __restrict__ Enorm, const float* __restrict__ in_w,
              const float* __restrict__ in_b, float* __restrict__ pooled)
{
    const int wid   = threadIdx.x >> 5, lane = threadIdx.x & 31;
    const int btile = blockIdx.x >> 2;
    const int h     = blockIdx.x & 3;
    const int dtile = wid * 16;
    const int m     = lane & 15, half = lane >> 4;
    const float* Arow = Enorm + (size_t)(btile * 16 + m) * (HH * DD) + h * DD;
    const float* Wrow = in_w + (size_t)(2 * DD + h * DH + dtile + m) * DD;  // Wv
    const int ko = half * 16;
    v8f c = {};
    for (int kb = 0; kb < DD; kb += 32) {
        v16bf a, bf;
        #pragma unroll
        for (int i = 0; i < 8; ++i) {
            a[i]     = (__bf16)Arow[kb + ko + i];
            a[8 + i] = (__bf16)Arow[kb + ko + 8 + i];
        }
        #pragma unroll
        for (int i = 0; i < 16; ++i) bf[i] = (__bf16)Wrow[kb + ko + i];
        c = __builtin_amdgcn_wmma_f32_16x16x32_bf16(false, a, false, bf,
                                                    (short)0, c, false, false);
    }
    const int dcol  = h * DH + dtile + m;
    const float bia = in_b[2 * DD + dcol];                 // bv
    const int rbase = btile * 16 + half * 8;
    #pragma unroll
    for (int r = 0; r < 8; ++r)
        pooled[(size_t)(rbase + r) * DD + dcol] = c[r] + bia;
}

// ---------------------------------------------------------------------------
// Kernel 4/5: generic C[M,N] = A[M,256] @ W[N,256]^T + bias, bf16 WMMA.
// ---------------------------------------------------------------------------
__global__ void __launch_bounds__(128)
gemm_bf16_kernel(const float* __restrict__ A, const float* __restrict__ W,
                 const float* __restrict__ bias, float* __restrict__ C, int N)
{
    const int wid    = threadIdx.x >> 5, lane = threadIdx.x & 31;
    const int tile   = blockIdx.x * 4 + wid;
    const int ntiles = N >> 4;
    const int mt     = tile / ntiles, nt = tile % ntiles;
    const int m      = lane & 15, half = lane >> 4;
    const float* Arow = A + (size_t)(mt * 16 + m) * DD;
    const float* Wrow = W + (size_t)(nt * 16 + m) * DD;
    const int ko = half * 16;
    v8f c = {};
    for (int kb = 0; kb < DD; kb += 32) {
        v16bf a, bf;
        #pragma unroll
        for (int i = 0; i < 8; ++i) {
            a[i]     = (__bf16)Arow[kb + ko + i];
            a[8 + i] = (__bf16)Arow[kb + ko + 8 + i];
        }
        #pragma unroll
        for (int i = 0; i < 16; ++i) bf[i] = (__bf16)Wrow[kb + ko + i];
        c = __builtin_amdgcn_wmma_f32_16x16x32_bf16(false, a, false, bf,
                                                    (short)0, c, false, false);
    }
    const int col   = nt * 16 + m;
    const float bia = bias[col];
    const int rbase = mt * 16 + half * 8;
    #pragma unroll
    for (int r = 0; r < 8; ++r)
        C[(size_t)(rbase + r) * N + col] = c[r] + bia;
}

// ---------------------------------------------------------------------------
extern "C" void kernel_launch(void* const* d_in, const int* in_sizes, int n_in,
                              void* d_out, int out_size, void* d_ws, size_t ws_size,
                              hipStream_t stream)
{
    const float* node_feat = (const float*)d_in[0];
    const int*   seg_ids   = (const int*)  d_in[1];
    const float* gq        = (const float*)d_in[2];
    const float* in_w      = (const float*)d_in[3];   // (3*256, 256)
    const float* in_b      = (const float*)d_in[4];
    const float* op_w      = (const float*)d_in[5];   // (256, 256)
    const float* op_b      = (const float*)d_in[6];
    const float* out_w     = (const float*)d_in[7];   // (512, 256)
    const float* out_b     = (const float*)d_in[8];
    float* out = (float*)d_out;                       // (256, 512) fp32

    // workspace layout (floats): qk | sbias | Enorm | pooled | attn  (~1.6 MB)
    float* ws     = (float*)d_ws;
    float* qk     = ws;                               // 1024
    float* sb     = ws + 1024;                        // 4
    float* Enorm  = ws + 2048;                        // BB*HH*DD = 262144
    float* pooled = Enorm + (size_t)BB * HH * DD;     // BB*DD = 65536
    float* attn   = pooled + (size_t)BB * DD;         // BB*DD = 65536

    setup_kernel<<<1, 256, 0, stream>>>(gq, in_w, in_b, qk, sb);
    segment_kernel<<<BB, 256, 0, stream>>>(node_feat, seg_ids, qk, sb, Enorm);
    pooled_kernel<<<(BB / 16) * HH, 128, 0, stream>>>(Enorm, in_w, in_b, pooled);
    gemm_bf16_kernel<<<(BB / 16) * (DD / 16) / 4, 128, 0, stream>>>(pooled, op_w, op_b, attn, DD);
    gemm_bf16_kernel<<<(BB / 16) * (OUTD / 16) / 4, 128, 0, stream>>>(attn, out_w, out_b, out, OUTD);
}
